// MessageMaxSelector_76295799046215
// MI455X (gfx1250) — matrix operand order repaired and checked by reading, not compiled, not run on hardware
//
#include <hip/hip_runtime.h>
#include <math.h>

// ---------------------------------------------------------------------------
// MI455X (gfx1250, wave32). bf16x3 split-precision GEMM pipeline:
// operands are pre-split ONCE into bf16 hi/lo planes; the GEMM hot loop is
// pure LDS traffic + v_wmma_f32_16x16x32_bf16 (3 WMMAs per 16x16x32 product).
// ---------------------------------------------------------------------------

typedef __attribute__((ext_vector_type(16))) __bf16 v16bf;
typedef __attribute__((ext_vector_type(8)))  __bf16 v8bf;
typedef __attribute__((ext_vector_type(8)))  float  v8f;

#define TILE_M 64
#define TILE_N 64
#define TILE_K 32
#define LDSK   40   // padded k-stride in bf16 units (80B, multiple of 16B)

#define CAUCHY_EXP_F 1.0100501670841680f   // exp(0.01)
#define PI_F 3.14159265358979323846f

// ---------------------------------------------------------------------------
// Generic strided batched GEMM on pre-split bf16 hi/lo operands.
//   C[b,m,n] = sum_k (Ah+Al)[b,m,k] * (Bh+Bl)[b,k,n]   (bf16x3: hh + hl + lh)
// A element: Ah/Al + b*bsA + m*sAm + k*sAk
// B element: Bh/Bl + b*bsB + k*sBk + n*sBn
// Outputs (element units, n contiguous, row stride ldC, batch stride bsC):
//   if bias  != null : v += bias[m]
//   if resid != null : v = 0.5*(v + resid_f32[b,m,n])
//   if Cf != null    : f32 store
//   if Ch != null    : hi/lo bf16 stores (for feeding the next GEMM)
// ---------------------------------------------------------------------------
__global__ void __launch_bounds__(128)
wmma_gemm_bf3(const __bf16* __restrict__ Ah, const __bf16* __restrict__ Al,
              long sAm, long sAk, long bsA,
              const __bf16* __restrict__ Bh, const __bf16* __restrict__ Bl,
              long sBk, long sBn, long bsB,
              int M, int N, int K,
              float* __restrict__ Cf, __bf16* __restrict__ Ch, __bf16* __restrict__ Cl,
              long ldC, long bsC,
              const float* __restrict__ bias,
              const float* __restrict__ resid, long bsR)
{
    __shared__ __bf16 lAh[TILE_M * LDSK], lAl[TILE_M * LDSK];   // [m][k]
    __shared__ __bf16 lBh[TILE_N * LDSK], lBl[TILE_N * LDSK];   // transposed [n][k]

    const int tid     = threadIdx.x;
    const int lane    = tid & 31;
    const int wid     = tid >> 5;
    const int waveM   = wid >> 1;
    const int waveN   = wid & 1;
    const int laneLow = lane & 15;
    const int laneHi  = lane >> 4;

    const int  bIdx = blockIdx.z;
    const long m0   = (long)blockIdx.y * TILE_M;
    const long n0   = (long)blockIdx.x * TILE_N;

    const __bf16* Abh = Ah + (long)bIdx * bsA;
    const __bf16* Abl = Al + (long)bIdx * bsA;
    const __bf16* Bbh = Bh + (long)bIdx * bsB;
    const __bf16* Bbl = Bl + (long)bIdx * bsB;

    v8f acc[2][2] = {};

    const int lm  = tid >> 1;          // 0..63  A row m
    const int lkh = (tid & 1) * 16;    // 0/16   A k half
    const int lbk = tid >> 2;          // 0..31  B row k
    const int lbn = (tid & 3) * 16;    // 0..48  B n quarter

    for (int k0 = 0; k0 < K; k0 += TILE_K) {
        // ---- A tile (64x32 bf16, hi+lo) ----
        {
            const long off = (m0 + lm) * sAm + (long)(k0 + lkh) * sAk;
            __bf16* dh = &lAh[lm * LDSK + lkh];
            __bf16* dl = &lAl[lm * LDSK + lkh];
            if (sAk == 1) {
                const v8bf* s8h = reinterpret_cast<const v8bf*>(Abh + off);
                const v8bf* s8l = reinterpret_cast<const v8bf*>(Abl + off);
                reinterpret_cast<v8bf*>(dh)[0] = s8h[0];
                reinterpret_cast<v8bf*>(dh)[1] = s8h[1];
                reinterpret_cast<v8bf*>(dl)[0] = s8l[0];
                reinterpret_cast<v8bf*>(dl)[1] = s8l[1];
            } else {
#pragma unroll
                for (int j = 0; j < 16; ++j) {
                    dh[j] = Abh[off + (long)j * sAk];
                    dl[j] = Abl[off + (long)j * sAk];
                }
            }
        }
        // ---- B tile (32x64 bf16, hi+lo) stored transposed [n][k] ----
        {
            const long off = (long)(k0 + lbk) * sBk + (n0 + lbn) * sBn;
            if (sBn == 1) {
                v8bf vh0 = reinterpret_cast<const v8bf*>(Bbh + off)[0];
                v8bf vh1 = reinterpret_cast<const v8bf*>(Bbh + off)[1];
                v8bf vl0 = reinterpret_cast<const v8bf*>(Bbl + off)[0];
                v8bf vl1 = reinterpret_cast<const v8bf*>(Bbl + off)[1];
#pragma unroll
                for (int j = 0; j < 8; ++j) {
                    lBh[(lbn + j) * LDSK + lbk]     = vh0[j];
                    lBh[(lbn + 8 + j) * LDSK + lbk] = vh1[j];
                    lBl[(lbn + j) * LDSK + lbk]     = vl0[j];
                    lBl[(lbn + 8 + j) * LDSK + lbk] = vl1[j];
                }
            } else {
#pragma unroll
                for (int j = 0; j < 16; ++j) {
                    lBh[(lbn + j) * LDSK + lbk] = Bbh[off + (long)j * sBn];
                    lBl[(lbn + j) * LDSK + lbk] = Bbl[off + (long)j * sBn];
                }
            }
        }
        __syncthreads();

        // ---- fragments: pure LDS copies (contiguous 8x bf16 -> b128) ----
        v16bf ah[2], al[2], bh[2], bl[2];
#pragma unroll
        for (int mi = 0; mi < 2; ++mi) {
            const int base = (waveM * 32 + mi * 16 + laneLow) * LDSK;
#pragma unroll
            for (int e = 0; e < 16; ++e) {
                // A 16x32: k = laneHi*16 + e (16 contiguous)
                int k = (laneHi << 4) + e;
                ah[mi][e] = lAh[base + k];
                al[mi][e] = lAl[base + k];
            }
        }
#pragma unroll
        for (int ni = 0; ni < 2; ++ni) {
            const int base = (waveN * 32 + ni * 16 + laneLow) * LDSK;
#pragma unroll
            for (int e = 0; e < 16; ++e) {
                // B 32x16: k = laneHi*16 + e (16 contiguous)
                int k = (laneHi << 4) + e;
                bh[ni][e] = lBh[base + k];
                bl[ni][e] = lBl[base + k];
            }
        }

        // ---- bf16x3 WMMA ----
#pragma unroll
        for (int mi = 0; mi < 2; ++mi)
#pragma unroll
            for (int ni = 0; ni < 2; ++ni) {
                acc[mi][ni] = __builtin_amdgcn_wmma_f32_16x16x32_bf16(
                    false, ah[mi], false, bh[ni], (short)0, acc[mi][ni], false, false);
                acc[mi][ni] = __builtin_amdgcn_wmma_f32_16x16x32_bf16(
                    false, ah[mi], false, bl[ni], (short)0, acc[mi][ni], false, false);
                acc[mi][ni] = __builtin_amdgcn_wmma_f32_16x16x32_bf16(
                    false, al[mi], false, bh[ni], (short)0, acc[mi][ni], false, false);
            }
        __syncthreads();
    }

    // ---- epilogue (C layout: n = lane&15, m = laneHi*8 + r) ----
    const long cb = (long)bIdx * bsC;
    const float* Rb = resid ? (resid + (long)bIdx * bsR) : nullptr;
#pragma unroll
    for (int mi = 0; mi < 2; ++mi)
#pragma unroll
        for (int ni = 0; ni < 2; ++ni) {
            long n = n0 + waveN * 32 + ni * 16 + laneLow;
#pragma unroll
            for (int r = 0; r < 8; ++r) {
                long m = m0 + waveM * 32 + mi * 16 + laneHi * 8 + r;
                float v = acc[mi][ni][r];
                if (bias)  v += bias[m];
                if (Rb)    v = 0.5f * (v + Rb[m * ldC + n]);
                long idx = cb + m * ldC + n;
                if (Cf) Cf[idx] = v;
                if (Ch) {
                    __bf16 h = (__bf16)v;
                    Ch[idx] = h;
                    Cl[idx] = (__bf16)(v - (float)h);
                }
            }
        }
}

// f32 -> (hi, lo) bf16 planes, one pass
__global__ void __launch_bounds__(256)
split_bf_kernel(const float* __restrict__ in, __bf16* __restrict__ h,
                __bf16* __restrict__ l, size_t n)
{
    size_t stride = (size_t)gridDim.x * 256;
    for (size_t i = (size_t)blockIdx.x * 256 + threadIdx.x; i < n; i += stride) {
        float f = in[i];
        __bf16 hh = (__bf16)f;
        h[i] = hh;
        l[i] = (__bf16)(f - (float)hh);
    }
}

// norms[b*R + r] = sum_k ((float)h + (float)l)^2 over strided access
__global__ void __launch_bounds__(256)
sumsq_pair_kernel(const __bf16* __restrict__ h, const __bf16* __restrict__ l,
                  float* __restrict__ out,
                  int count, long kStride, long rStride, long bStride, int R)
{
    int row = blockIdx.x;
    int b = row / R, r = row % R;
    long base = (long)b * bStride + (long)r * rStride;
    float s = 0.f;
    for (int k = threadIdx.x; k < count; k += 256) {
        long i = base + (long)k * kStride;
        float v = (float)h[i] + (float)l[i];
        s += v * v;
    }
    __shared__ float red[256];
    red[threadIdx.x] = s;
    __syncthreads();
    for (int off = 128; off > 0; off >>= 1) {
        if (threadIdx.x < off) red[threadIdx.x] += red[threadIdx.x + off];
        __syncthreads();
    }
    if (threadIdx.x == 0) out[row] = red[0];
}

// Fused cauchy -> relu -> softmax along the moving axis; writes normalized
// result as bf16 hi/lo planes (inputs S f32 used as scratch for pass 1).
__global__ void __launch_bounds__(256)
cauchy_softmax_kernel(float* __restrict__ S,
                      const float* __restrict__ normM,
                      const float* __restrict__ normF,
                      int axisLen, int fixedCount, long elemStride, long fixedStride,
                      __bf16* __restrict__ Oh, __bf16* __restrict__ Ol)
{
    int idx = blockIdx.x;                 // b*fixedCount + f
    int b = idx / fixedCount, f = idx % fixedCount;
    long bbase = (long)b * axisLen * (long)fixedCount + (long)f * fixedStride;
    float* base = S + bbase;
    const float nfv = normF[(long)b * fixedCount + f];
    const float* nmv = normM + (long)b * axisLen;

    float lsum = 0.f;
    for (int i = threadIdx.x; i < axisLen; i += 256) {
        float s = base[(long)i * elemStride];
        float q = (s * s) / (nmv[i] * nfv);
        float v = powf(q, CAUCHY_EXP_F) - (i == f ? 1.f : 0.f);
        float c = 0.5f - 0.5f * cosf(PI_F * v);
        float e = expf(fmaxf(c, 0.f));    // values in [0,1] -> safe w/o max-sub
        base[(long)i * elemStride] = e;
        lsum += e;
    }
    __shared__ float red[256];
    red[threadIdx.x] = lsum;
    __syncthreads();
    for (int off = 128; off > 0; off >>= 1) {
        if (threadIdx.x < off) red[threadIdx.x] += red[threadIdx.x + off];
        __syncthreads();
    }
    float inv = 1.f / red[0];
    for (int i = threadIdx.x; i < axisLen; i += 256) {
        long e = bbase + (long)i * elemStride;
        float v = S[e] * inv;
        __bf16 h = (__bf16)v;
        Oh[e] = h;
        Ol[e] = (__bf16)(v - (float)h);
    }
}

extern "C" void kernel_launch(void* const* d_in, const int* in_sizes, int n_in,
                              void* d_out, int out_size, void* d_ws, size_t ws_size,
                              hipStream_t stream) {
    (void)in_sizes; (void)n_in; (void)out_size; (void)ws_size;
    const int Bn = 8, D = 1024, L = 2048;
    const long DL = (long)D * L;
    const long DD = (long)D * D;
    const long LL = (long)L * L;
    const size_t nBDL = (size_t)Bn * DL;

    const float* x = (const float*)d_in[0];
    const float* W[4]  = {(const float*)d_in[1], (const float*)d_in[3],
                          (const float*)d_in[5], (const float*)d_in[7]};
    const float* bv[4] = {(const float*)d_in[2], (const float*)d_in[4],
                          (const float*)d_in[6], (const float*)d_in[8]};
    float* y = (float*)d_out;

    // ---- workspace layout ----
    __bf16* p = (__bf16*)d_ws;
    __bf16* xh = p;            p += nBDL;
    __bf16* xl = p;            p += nBDL;
    __bf16* Wh = p;            p += (size_t)12 * DD;   // 4 stacks x 3 layers
    __bf16* Wl = p;            p += (size_t)12 * DD;
    __bf16* b0h = p;           p += nBDL;
    __bf16* b0l = p;           p += nBDL;
    __bf16* b1h = p;           p += nBDL;
    __bf16* b1l = p;           p += nBDL;
    __bf16* sxh = p;           p += nBDL;
    __bf16* sxl = p;           p += nBDL;
    __bf16* cxh = p;           p += nBDL;
    __bf16* cxl = p;           p += nBDL;
    __bf16* dsh = p;           p += (size_t)Bn * DD;
    __bf16* dsl = p;           p += (size_t)Bn * DD;
    __bf16* lsh = p;           p += (size_t)Bn * LL;
    __bf16* lsl = p;           p += (size_t)Bn * LL;
    float* Sf = (float*)p;                              // raw scores scratch
    float* nm = Sf + (size_t)Bn * LL;
    float* nf = nm + (size_t)Bn * L;

    dim3 blk(128);
    auto gemm = [&](const __bf16* Ah, const __bf16* Al, long sAm, long sAk, long bsA,
                    const __bf16* Bh, const __bf16* Bl, long sBk, long sBn, long bsB,
                    int M, int N, int K,
                    float* Cf, __bf16* Ch, __bf16* Cl, long ldC, long bsC,
                    const float* bias, const float* resid, long bsR) {
        dim3 grid((unsigned)(N / TILE_N), (unsigned)(M / TILE_M), (unsigned)Bn);
        hipLaunchKernelGGL(wmma_gemm_bf3, grid, blk, 0, stream,
                           Ah, Al, sAm, sAk, bsA, Bh, Bl, sBk, sBn, bsB,
                           M, N, K, Cf, Ch, Cl, ldC, bsC, bias, resid, bsR);
    };
    auto split = [&](const float* in, __bf16* h, __bf16* l, size_t n) {
        hipLaunchKernelGGL(split_bf_kernel, dim3(4096), dim3(256), 0, stream, in, h, l, n);
    };

    // ---- one-time splits ----
    split(x, xh, xl, nBDL);
    for (int i = 0; i < 4; ++i)
        split(W[i], Wh + (size_t)i * 3 * DD, Wl + (size_t)i * 3 * DD, (size_t)3 * DD);

    // 3-layer pointwise MLP: W(D,D) x in(B,D,L) + bias, bf16 h/l chained
    auto mlp = [&](int wi, __bf16* dsth, __bf16* dstl) {
        const __bf16* wsh = Wh + (size_t)wi * 3 * DD;
        const __bf16* wsl = Wl + (size_t)wi * 3 * DD;
        gemm(wsh + 0 * DD, wsl + 0 * DD, D, 1, 0, xh,  xl,  L, 1, DL, D, L, D,
             nullptr, b0h, b0l, L, DL, bv[wi] + 0 * D, nullptr, 0);
        gemm(wsh + 1 * DD, wsl + 1 * DD, D, 1, 0, b0h, b0l, L, 1, DL, D, L, D,
             nullptr, b1h, b1l, L, DL, bv[wi] + 1 * D, nullptr, 0);
        gemm(wsh + 2 * DD, wsl + 2 * DD, D, 1, 0, b1h, b1l, L, 1, DL, D, L, D,
             nullptr, dsth, dstl, L, DL, bv[wi] + 2 * D, nullptr, 0);
    };

    // ---- channel-mixing branch: dscore (B,D,D), softmax over m ----
    mlp(0, sxh, sxl);   // dsx
    mlp(1, cxh, cxl);   // dcx
    hipLaunchKernelGGL(sumsq_pair_kernel, dim3(Bn * D), dim3(256), 0, stream,
                       sxh, sxl, nm, L, 1L, (long)L, DL, D);
    hipLaunchKernelGGL(sumsq_pair_kernel, dim3(Bn * D), dim3(256), 0, stream,
                       cxh, cxl, nf, L, 1L, (long)L, DL, D);
    // S = dsx . dcx^T  (NT, K=L) -> Sf (f32)
    gemm(sxh, sxl, L, 1, DL, cxh, cxl, 1, (long)L, DL, D, D, L,
         Sf, nullptr, nullptr, D, DD, nullptr, nullptr, 0);
    hipLaunchKernelGGL(cauchy_softmax_kernel, dim3(Bn * D), dim3(256), 0, stream,
                       Sf, nm, nf, D, D, (long)D, 1L, dsh, dsl);

    // ---- token-mixing branch: lscore (B,L,L), softmax over n ----
    mlp(2, sxh, sxl);   // lsx
    mlp(3, cxh, cxl);   // lcx
    hipLaunchKernelGGL(sumsq_pair_kernel, dim3(Bn * L), dim3(256), 0, stream,
                       sxh, sxl, nm, D, (long)L, 1L, DL, L);
    hipLaunchKernelGGL(sumsq_pair_kernel, dim3(Bn * L), dim3(256), 0, stream,
                       cxh, cxl, nf, D, (long)L, 1L, DL, L);
    // S2 = lsx^T . lcx  (TN, K=D) -> Sf (f32)
    gemm(sxh, sxl, 1, (long)L, DL, cxh, cxl, L, 1, DL, L, L, D,
         Sf, nullptr, nullptr, L, LL, nullptr, nullptr, 0);
    hipLaunchKernelGGL(cauchy_softmax_kernel, dim3(Bn * L), dim3(256), 0, stream,
                       Sf, nf, nm, L, L, 1L, (long)L, lsh, lsl);

    // ---- combine: y = (dscore @ x @ lscore + x) / 2 ----
    // t1 = dscore @ x -> b0 (h/l, reuse)
    gemm(dsh, dsl, D, 1, DD, xh, xl, L, 1, DL, D, L, D,
         nullptr, b0h, b0l, L, DL, nullptr, nullptr, 0);
    // y = 0.5*(t1 @ lscore + x), exact f32 residual
    gemm(b0h, b0l, L, 1, DL, lsh, lsl, L, 1, LL, D, L, L,
         y, nullptr, nullptr, L, DL, nullptr, x, DL);
}
